// PoseVQVAE_27109833573047
// MI455X (gfx1250) — matrix operand order repaired and checked by reading, not compiled, not run on hardware
//
#include <hip/hip_runtime.h>
#include <hip/hip_bf16.h>

// ---------------------------------------------------------------------------
// Types for CDNA5 WMMA
// ---------------------------------------------------------------------------
typedef __attribute__((ext_vector_type(16))) __bf16 v16bf;
typedef __attribute__((ext_vector_type(8)))  float  v8f;

union FragBF {
    v16bf v;
    uint4 u[2];
};

// ---------------------------------------------------------------------------
// Problem constants
// ---------------------------------------------------------------------------
#define BATCH    131072
#define FRAME    267
#define H1       512
#define LATENT   256
#define NCODE    1024
#define ENC_IN   534          // FRAME*2
#define ENC_PAD  544          // 17 * 32
#define DEC_IN   523          // LATENT + FRAME
#define DEC_PAD  544
#define TILE_B   64

// K-step counts (K/32) and N-tile counts (N/16)
#define KT_ENC   17
#define KT_H     16
#define KT_LAT   8
#define NT_H     32
#define NT_MU    16
#define NT_E     64
#define NT_O     17           // ceil(267/16)

// ---------------------------------------------------------------------------
// bf16 helpers (round-to-nearest-even)
// ---------------------------------------------------------------------------
__device__ __host__ inline unsigned short f2bf(float f) {
    unsigned u = __builtin_bit_cast(unsigned, f);
    u += 0x7FFFu + ((u >> 16) & 1u);
    return (unsigned short)(u >> 16);
}
__device__ inline float bf2f(unsigned short h) {
    unsigned u = ((unsigned)h) << 16;
    return __builtin_bit_cast(float, u);
}

// ---------------------------------------------------------------------------
// Weight repack: W [Kin x Nout] f32 row-major  ->  bf16 WMMA B-fragments.
// Fragment order: frag = ntile*Ktiles + ktile (so the K loop streams
// contiguously). Per ISA B layout (32x16, 16-bit): lane n holds column
// ntile*16 + (lane&15); elems 0..7 hold K = kb..kb+7, elems 8..15 hold
// K = kb+16..kb+23, kb = ktile*32 + (lane<16 ? 0 : 8). One thread per (frag, lane): 32B store.
// ---------------------------------------------------------------------------
__global__ void pack_weight(const float* __restrict__ W,
                            unsigned short* __restrict__ P,
                            int Kin, int Nout, int Ktiles, int Ntiles) {
    int t = blockIdx.x * blockDim.x + threadIdx.x;
    int total = Ktiles * Ntiles * 32;
    if (t >= total) return;
    int lane = t & 31;
    int frag = t >> 5;
    int ktile = frag % Ktiles;
    int ntile = frag / Ktiles;
    int n  = ntile * 16 + (lane & 15);
    int kb = ktile * 32 + ((lane < 16) ? 0 : 8);
    unsigned words[8];
#pragma unroll
    for (int j = 0; j < 8; ++j) {
        int k0 = kb + 2 * j + ((j >= 4) ? 8 : 0);
        int k1 = k0 + 1;
        float f0 = (k0 < Kin && n < Nout) ? W[(size_t)k0 * Nout + n] : 0.0f;
        float f1 = (k1 < Kin && n < Nout) ? W[(size_t)k1 * Nout + n] : 0.0f;
        words[j] = (unsigned)f2bf(f0) | ((unsigned)f2bf(f1) << 16);
    }
    uint4* dst = (uint4*)(P + (size_t)t * 16);
    dst[0] = make_uint4(words[0], words[1], words[2], words[3]);
    dst[1] = make_uint4(words[4], words[5], words[6], words[7]);
}

// ---------------------------------------------------------------------------
// ||e_k||^2 for the VQ distance. embed is [LATENT x NCODE] row-major.
// ---------------------------------------------------------------------------
__global__ void vq_enorm(const float* __restrict__ embed,
                         float* __restrict__ enorm) {
    int k = blockIdx.x * blockDim.x + threadIdx.x;
    if (k >= NCODE) return;
    float s = 0.0f;
    for (int j = 0; j < LATENT; ++j) {
        float e = embed[(size_t)j * NCODE + k];
        s += e * e;
    }
    enorm[k] = s;
}

__global__ void vq_init(unsigned* __restrict__ hist, float* __restrict__ lossAcc) {
    int t = blockIdx.x * blockDim.x + threadIdx.x;
    if (t < NCODE) hist[t] = 0u;
    if (t == 0) *lossAcc = 0.0f;
}

// ---------------------------------------------------------------------------
// Tiled GEMM layer over one 64-row activation tile (all 8 waves cooperate).
//   in  : LDS bf16 [64 x inStride], logical K = Ksteps*32 (zero padded)
//   Wp  : packed bf16 fragments in global (L2-resident)
//   out : LDS bf16 [64 x outStride], bias + optional ReLU applied
// A-fragment layout (16-bit A 16x32): lane row = lane&15; elems 0..7 hold
// K = kb..kb+7, elems 8..15 hold K = kb+16..kb+23 with kb = (lane<16?0:8).
// ---------------------------------------------------------------------------
__device__ __forceinline__ void gemm_layer(
    const unsigned short* __restrict__ in, int inStride, int Ksteps,
    const unsigned short* __restrict__ Wp, const float* __restrict__ bias,
    int Ntiles, unsigned short* __restrict__ out, int outStride, bool relu,
    int lane, int wave) {
    const int mrow = lane & 15;
    const int ahi  = (lane < 16) ? 0 : 8;      // A: K sub-offset
    const int chi  = (lane < 16) ? 0 : 8;      // C/D: row offset
    const int nFrags = 4 * Ntiles;
    for (int f = wave; f < nFrags; f += 8) {
        int mt = f & 3;
        int nt = f >> 2;
        v8f acc = {};
        const unsigned short* arow = in + (size_t)(mt * 16 + mrow) * inStride + ahi;
        const uint4* wptr = (const uint4*)Wp + ((size_t)(nt * Ksteps) * 32 + lane) * 2;
        for (int k = 0; k < Ksteps; ++k) {
            FragBF a, b;
            a.u[0] = *(const uint4*)(arow);
            a.u[1] = *(const uint4*)(arow + 16);
            b.u[0] = wptr[0];
            b.u[1] = wptr[1];
            arow += 32;
            wptr += 64;
            acc = __builtin_amdgcn_wmma_f32_16x16x32_bf16(
                false, a.v, false, b.v, (short)0, acc, false, false);
        }
        int coln = nt * 16 + (lane & 15);
        float bv = bias[coln];
#pragma unroll
        for (int i = 0; i < 8; ++i) {
            float v = acc[i] + bv;
            if (relu) v = fmaxf(v, 0.0f);
            out[(size_t)(mt * 16 + i + chi) * outStride + coln] = f2bf(v);
        }
    }
}

// ---------------------------------------------------------------------------
// Fully fused VQ-VAE forward for a 64-row batch tile.
// ---------------------------------------------------------------------------
__global__ __launch_bounds__(256) void vqvae_fused(
    const float* __restrict__ x,  const float* __restrict__ c,
    const unsigned short* __restrict__ P1, const float* __restrict__ b1,
    const unsigned short* __restrict__ P2, const float* __restrict__ b2,
    const unsigned short* __restrict__ P3, const float* __restrict__ b3,
    const unsigned short* __restrict__ Pmu, const float* __restrict__ bmu,
    const unsigned short* __restrict__ Pe,  const float* __restrict__ enorm,
    const float* __restrict__ embed,
    const unsigned short* __restrict__ P4, const float* __restrict__ b4,
    const unsigned short* __restrict__ P5, const float* __restrict__ b5,
    const unsigned short* __restrict__ P6, const float* __restrict__ b6,
    const unsigned short* __restrict__ Po, const float* __restrict__ bo,
    float* __restrict__ out, float* __restrict__ lossAcc,
    unsigned* __restrict__ hist) {

    extern __shared__ char smem[];
    unsigned short* A  = (unsigned short*)(smem);                 // 64 x 544 bf16
    unsigned short* Bf = (unsigned short*)(smem + 69632);         // 64 x 512 bf16
    float*          vqv = (float*)(smem + 135168);                // 64 x 32
    int*            vqi = (int*)  (smem + 143360);                // 64 x 32
    int*            ind = (int*)  (smem + 151552);                // 64
    float*          red = (float*)(smem + 151808);                // 8

    const int tid  = threadIdx.x;
    const int lane = tid & 31;
    const int wave = tid >> 5;
    const size_t rowBase = (size_t)blockIdx.x * TILE_B;

    // ---------------- load s = [x | c] (bf16, zero-padded to 544) ----------
    for (int i = tid; i < TILE_B * ENC_PAD; i += 256) {
        int r = i / ENC_PAD, col = i % ENC_PAD;
        float f = 0.0f;
        if (col < FRAME)            f = x[(rowBase + r) * FRAME + col];
        else if (col < 2 * FRAME)   f = c[(rowBase + r) * FRAME + (col - FRAME)];
        A[(size_t)r * ENC_PAD + col] = f2bf(f);
    }
    __syncthreads();

    // ---------------- encoder ----------------
    gemm_layer(A, ENC_PAD, KT_ENC, P1, b1, NT_H, Bf, H1, true, lane, wave);
    __syncthreads();
    gemm_layer(Bf, H1, KT_H, P2, b2, NT_H, A, ENC_PAD, true, lane, wave);
    __syncthreads();
    gemm_layer(A, ENC_PAD, KT_H, P3, b3, NT_H, Bf, H1, true, lane, wave);
    __syncthreads();
    gemm_layer(Bf, H1, KT_H, Pmu, bmu, NT_MU, A, ENC_PAD, false, lane, wave); // mu
    __syncthreads();

    // ---------------- VQ: argmin_k ( ||e_k||^2 - 2 mu.e_k ) ---------------
    {
        const int mrow = lane & 15;
        const int ahi  = (lane < 16) ? 0 : 8;
        const int chi  = (lane < 16) ? 0 : 8;
        int mt   = wave & 3;
        int half = wave >> 2;      // column half: 0 -> codes 0..511, 1 -> 512..1023
        float minv[8];
        int   mini[8];
#pragma unroll
        for (int i = 0; i < 8; ++i) { minv[i] = 3.4e38f; mini[i] = 0; }
        const unsigned short* arow0 = A + (size_t)(mt * 16 + mrow) * ENC_PAD + ahi;
        for (int nt = half * 32; nt < half * 32 + 32; ++nt) {
            v8f acc = {};
            const unsigned short* arow = arow0;
            const uint4* wptr = (const uint4*)Pe + ((size_t)(nt * KT_LAT) * 32 + lane) * 2;
            for (int k = 0; k < KT_LAT; ++k) {
                FragBF a, b;
                a.u[0] = *(const uint4*)(arow);
                a.u[1] = *(const uint4*)(arow + 16);
                b.u[0] = wptr[0];
                b.u[1] = wptr[1];
                arow += 32;
                wptr += 64;
                acc = __builtin_amdgcn_wmma_f32_16x16x32_bf16(
                    false, a.v, false, b.v, (short)0, acc, false, false);
            }
            int coln = nt * 16 + (lane & 15);
            float en = enorm[coln];
#pragma unroll
            for (int i = 0; i < 8; ++i) {
                float d = en - 2.0f * acc[i];
                if (d < minv[i]) { minv[i] = d; mini[i] = coln; }
            }
        }
#pragma unroll
        for (int i = 0; i < 8; ++i) {
            int r = mt * 16 + i + chi;
            int slot = (lane & 15) + 16 * half;
            vqv[r * 32 + slot] = minv[i];
            vqi[r * 32 + slot] = mini[i];
        }
    }
    __syncthreads();
    if (tid < TILE_B) {
        float bv = 3.4e38f;
        int bi = 0;
        for (int s = 0; s < 32; ++s) {
            float v = vqv[tid * 32 + s];
            if (v < bv) { bv = v; bi = vqi[tid * 32 + s]; }
        }
        ind[tid] = bi;
        atomicAdd(&hist[bi], 1u);
    }
    __syncthreads();

    // ------- gather quantize (overwrite mu), accumulate commitment loss ---
    float lsum = 0.0f;
    {
        int r  = tid >> 2;
        int c0 = (tid & 3) * 64;
        int id = ind[r];
        for (int j = 0; j < 64; ++j) {
            int col = c0 + j;
            float q = embed[(size_t)col * NCODE + id];
            float m = bf2f(A[(size_t)r * ENC_PAD + col]);
            float d = q - m;
            lsum += d * d;
            A[(size_t)r * ENC_PAD + col] = f2bf(q);
        }
    }
    // fill s2 tail: c in cols 256..522, zeros 523..543
    for (int i = tid; i < TILE_B * (DEC_PAD - LATENT); i += 256) {
        int r = i / (DEC_PAD - LATENT);
        int col = LATENT + i % (DEC_PAD - LATENT);
        float f = (col < DEC_IN) ? c[(rowBase + r) * FRAME + (col - LATENT)] : 0.0f;
        A[(size_t)r * ENC_PAD + col] = f2bf(f);
    }
    // reduce loss across block -> one atomic
    for (int off = 16; off; off >>= 1) lsum += __shfl_down(lsum, off, 32);
    if (lane == 0) red[wave] = lsum;
    __syncthreads();
    if (tid == 0) {
        float t = 0.0f;
        for (int i = 0; i < 8; ++i) t += red[i];
        atomicAdd(lossAcc, t);
    }
    __syncthreads();

    // ---------------- decoder ----------------
    gemm_layer(A, ENC_PAD, KT_ENC, P4, b4, NT_H, Bf, H1, true, lane, wave);
    __syncthreads();
    gemm_layer(Bf, H1, KT_H, P5, b5, NT_H, A, ENC_PAD, true, lane, wave);
    __syncthreads();
    gemm_layer(A, ENC_PAD, KT_H, P6, b6, NT_H, Bf, H1, true, lane, wave);
    __syncthreads();

    // ---------------- output layer: recon -> global f32 -------------------
    {
        const int mrow = lane & 15;
        const int ahi  = (lane < 16) ? 0 : 8;
        const int chi  = (lane < 16) ? 0 : 8;
        for (int f = wave; f < 4 * NT_O; f += 8) {
            int mt = f & 3;
            int nt = f >> 2;
            v8f acc = {};
            const unsigned short* arow = Bf + (size_t)(mt * 16 + mrow) * H1 + ahi;
            const uint4* wptr = (const uint4*)Po + ((size_t)(nt * KT_H) * 32 + lane) * 2;
            for (int k = 0; k < KT_H; ++k) {
                FragBF a, b;
                a.u[0] = *(const uint4*)(arow);
                a.u[1] = *(const uint4*)(arow + 16);
                b.u[0] = wptr[0];
                b.u[1] = wptr[1];
                arow += 32;
                wptr += 64;
                acc = __builtin_amdgcn_wmma_f32_16x16x32_bf16(
                    false, a.v, false, b.v, (short)0, acc, false, false);
            }
            int coln = nt * 16 + (lane & 15);
            if (coln < FRAME) {
                float bv = bo[coln];
#pragma unroll
                for (int i = 0; i < 8; ++i) {
                    size_t rg = rowBase + (size_t)(mt * 16 + i + chi);
                    out[rg * FRAME + coln] = acc[i] + bv;
                }
            }
        }
    }
}

// ---------------------------------------------------------------------------
// Scalars: loss mean + perplexity from histogram.
// ---------------------------------------------------------------------------
__global__ void vq_finalize(const unsigned* __restrict__ hist,
                            const float* __restrict__ lossAcc,
                            float* __restrict__ out, int outSize) {
    __shared__ float sred[32];
    int t = threadIdx.x;                       // 1024 threads
    float p = (float)hist[t] * (1.0f / (float)BATCH);
    float s = p * logf(p + 1e-10f);
    int lane = t & 31, wv = t >> 5;
    for (int off = 16; off; off >>= 1) s += __shfl_down(s, off, 32);
    if (lane == 0) sred[wv] = s;
    __syncthreads();
    if (t < 32) {
        float v = sred[t];
        for (int off = 16; off; off >>= 1) v += __shfl_down(v, off, 32);
        if (t == 0) {
            out[outSize - 2] = lossAcc[0] * (1.0f / ((float)BATCH * (float)LATENT));
            out[outSize - 1] = expf(-v);
        }
    }
}

// ---------------------------------------------------------------------------
// Host launcher
// ---------------------------------------------------------------------------
extern "C" void kernel_launch(void* const* d_in, const int* in_sizes, int n_in,
                              void* d_out, int out_size, void* d_ws, size_t ws_size,
                              hipStream_t stream) {
    (void)in_sizes; (void)n_in; (void)ws_size;
    const float* x    = (const float*)d_in[0];
    const float* c    = (const float*)d_in[1];
    const float* W1   = (const float*)d_in[2];
    const float* b1   = (const float*)d_in[3];
    const float* W2   = (const float*)d_in[4];
    const float* b2   = (const float*)d_in[5];
    const float* W3   = (const float*)d_in[6];
    const float* b3   = (const float*)d_in[7];
    const float* Wmu  = (const float*)d_in[8];
    const float* bmu  = (const float*)d_in[9];
    const float* W4   = (const float*)d_in[10];
    const float* b4   = (const float*)d_in[11];
    const float* W5   = (const float*)d_in[12];
    const float* b5   = (const float*)d_in[13];
    const float* W6   = (const float*)d_in[14];
    const float* b6   = (const float*)d_in[15];
    const float* Wo   = (const float*)d_in[16];
    const float* bo   = (const float*)d_in[17];
    const float* emb  = (const float*)d_in[18];

    // workspace layout (bytes, 256-aligned blocks)
    char* ws = (char*)d_ws;
    size_t off = 0;
    auto alloc = [&](size_t bytes) {
        char* p = ws + off;
        off += (bytes + 255) & ~size_t(255);
        return p;
    };
    unsigned short* P1  = (unsigned short*)alloc((size_t)KT_ENC * NT_H  * 512 * 2);
    unsigned short* P2  = (unsigned short*)alloc((size_t)KT_H   * NT_H  * 512 * 2);
    unsigned short* P3  = (unsigned short*)alloc((size_t)KT_H   * NT_H  * 512 * 2);
    unsigned short* Pmu = (unsigned short*)alloc((size_t)KT_H   * NT_MU * 512 * 2);
    unsigned short* Pe  = (unsigned short*)alloc((size_t)KT_LAT * NT_E  * 512 * 2);
    unsigned short* P4  = (unsigned short*)alloc((size_t)KT_ENC * NT_H  * 512 * 2);
    unsigned short* P5  = (unsigned short*)alloc((size_t)KT_H   * NT_H  * 512 * 2);
    unsigned short* P6  = (unsigned short*)alloc((size_t)KT_H   * NT_H  * 512 * 2);
    unsigned short* Po  = (unsigned short*)alloc((size_t)KT_H   * NT_O  * 512 * 2);
    float*    enorm   = (float*)alloc(NCODE * 4);
    unsigned* hist    = (unsigned*)alloc(NCODE * 4);
    float*    lossAcc = (float*)alloc(16);

    auto packLaunch = [&](const float* W, unsigned short* P, int Kin, int Nout,
                          int Kt, int Nt) {
        int total = Kt * Nt * 32;
        pack_weight<<<(total + 255) / 256, 256, 0, stream>>>(W, P, Kin, Nout, Kt, Nt);
    };
    packLaunch(W1,  P1,  ENC_IN, H1,     KT_ENC, NT_H);
    packLaunch(W2,  P2,  H1,     H1,     KT_H,   NT_H);
    packLaunch(W3,  P3,  H1,     H1,     KT_H,   NT_H);
    packLaunch(Wmu, Pmu, H1,     LATENT, KT_H,   NT_MU);
    packLaunch(emb, Pe,  LATENT, NCODE,  KT_LAT, NT_E);
    packLaunch(W4,  P4,  DEC_IN, H1,     KT_ENC, NT_H);
    packLaunch(W5,  P5,  H1,     H1,     KT_H,   NT_H);
    packLaunch(W6,  P6,  H1,     H1,     KT_H,   NT_H);
    packLaunch(Wo,  Po,  H1,     FRAME,  KT_H,   NT_O);

    vq_enorm<<<NCODE / 256, 256, 0, stream>>>(emb, enorm);
    vq_init<<<NCODE / 256, 256, 0, stream>>>(hist, lossAcc);

    const size_t smemBytes = 151872;   // A(69632)+B(65536)+vqv(8192)+vqi(8192)+ind(256)+red(64)
    vqvae_fused<<<BATCH / TILE_B, 256, smemBytes, stream>>>(
        x, c, P1, b1, P2, b2, P3, b3, Pmu, bmu, Pe, enorm, emb,
        P4, b4, P5, b5, P6, b6, Po, bo,
        (float*)d_out, lossAcc, hist);

    vq_finalize<<<1, NCODE, 0, stream>>>(hist, lossAcc, (float*)d_out, out_size);
}
